// MultiHeadAttention_9380208575152
// MI455X (gfx1250) — matrix operand order repaired and checked by reading, not compiled, not run on hardware
//
#include <hip/hip_runtime.h>
#include <stdint.h>

typedef __attribute__((ext_vector_type(16))) __bf16 bf16x16;
typedef __attribute__((ext_vector_type(8)))  float  f32x8;

__device__ __forceinline__ unsigned short f2bf(float f) {
  unsigned u = __builtin_bit_cast(unsigned, f);
  u = (u + 0x7FFFu + ((u >> 16) & 1u)) >> 16;   // round-to-nearest-even
  return (unsigned short)u;
}

union FragU { uint4 u[2]; bf16x16 v; };

// A-fragment (16xK, 32-wide K window) per ISA 7.12.2 16-bit A layout:
// lane holds row (lane&15); K chunks {(lane>>4)*8 ..+7} and {16+(lane>>4)*8 ..+7}.
__device__ __forceinline__ bf16x16 load_frag_a(const unsigned short* base, int stride, int lane) {
  const unsigned short* row = base + (lane & 15) * stride + ((lane >> 4) * 8);
  FragU f;
  f.u[0] = *(const uint4*)(row);
  f.u[1] = *(const uint4*)(row + 16);
  return f.v;
}

// B-fragment: lane holds column n = rowOff+(lane&15); K chunks colOff+{(lane>>4)*8 ..+7}
// and colOff+{16+(lane>>4)*8 ..+7}, gathered from row-major [n][k] LDS.
__device__ __forceinline__ bf16x16 load_frag_b(const unsigned short* base, int stride, int lane,
                                               int rowOff, int colOff) {
  const unsigned short* row =
      base + (size_t)(rowOff + (lane & 15)) * stride + colOff + ((lane >> 4) * 8);
  FragU f;
  f.u[0] = *(const uint4*)(row);
  f.u[1] = *(const uint4*)(row + 16);
  return f.v;
}

__device__ __forceinline__ f32x8 wmma_bf16(bf16x16 a, bf16x16 b, f32x8 c) {
  return __builtin_amdgcn_wmma_f32_16x16x32_bf16(false, a, false, b, (short)0, c, false, false);
}

// CDNA5 async global->LDS copy (16B per lane, ASYNCcnt-tracked, no VGPR data).
// VDST VGPR = per-lane LDS byte address, VADDR = 64-bit global address, no SADDR.
__device__ __forceinline__ void async_copy_b128(const unsigned short* gsrc,
                                                unsigned short* ldst) {
  unsigned lds_off = (unsigned)(size_t)ldst;              // LDS aperture: low 32 bits = offset
  unsigned long long gaddr = (unsigned long long)(size_t)gsrc;
  asm volatile("global_load_async_to_lds_b128 %0, %1, off"
               :: "v"(lds_off), "v"(gaddr)
               : "memory");
}

__device__ __forceinline__ void wait_async0() {
  asm volatile("s_wait_asynccnt 0x0" ::: "memory");
}

// ---------------------------------------------------------------------------
// Projection: Out = relu(X @ W^T + bias), X:(4096,1024) f32, W:(1024,1024) f32,
// Out bf16 (workspace). blockIdx.z selects q/k/v.
// ---------------------------------------------------------------------------
__global__ __launch_bounds__(256) void proj_relu_gemm(
    const float* __restrict__ Xq, const float* __restrict__ Xk, const float* __restrict__ Xv,
    const float* __restrict__ Wqm, const float* __restrict__ bqv,
    const float* __restrict__ Wkm, const float* __restrict__ bkv,
    const float* __restrict__ Wvm, const float* __restrict__ bvv,
    unsigned short* __restrict__ qp, unsigned short* __restrict__ kp,
    unsigned short* __restrict__ vp) {
  __shared__ unsigned short As[128 * 32];
  __shared__ unsigned short Bs[128 * 32];

  const float* X; const float* W; const float* bias; unsigned short* Out;
  if (blockIdx.z == 0)      { X = Xq; W = Wqm; bias = bqv; Out = qp; }
  else if (blockIdx.z == 1) { X = Xk; W = Wkm; bias = bkv; Out = kp; }
  else                      { X = Xv; W = Wvm; bias = bvv; Out = vp; }

  const int tid  = threadIdx.x;
  const int lane = tid & 31;
  const int wave = tid >> 5;
  const int m0 = blockIdx.x * 128;
  const int n0 = blockIdx.y * 128;
  const int wm = (wave & 3) * 32;   // wave M offset in tile
  const int wn = (wave >> 2) * 64;  // wave N offset in tile

  const f32x8 zf = {0.f, 0.f, 0.f, 0.f, 0.f, 0.f, 0.f, 0.f};
  f32x8 acc[2][4];
#pragma unroll
  for (int mt = 0; mt < 2; ++mt)
#pragma unroll
    for (int nt = 0; nt < 4; ++nt) acc[mt][nt] = zf;

  for (int k0 = 0; k0 < 1024; k0 += 32) {
    __syncthreads();
#pragma unroll
    for (int r = 0; r < 4; ++r) {
      int j = tid + 256 * r;
      int row = j >> 3;
      int c4 = (j & 7) * 4;
      float4 a = *(const float4*)(X + (size_t)(m0 + row) * 1024 + k0 + c4);
      float4 b = *(const float4*)(W + (size_t)(n0 + row) * 1024 + k0 + c4);
      uint2 pa, pb;
      pa.x = (unsigned)f2bf(a.x) | ((unsigned)f2bf(a.y) << 16);
      pa.y = (unsigned)f2bf(a.z) | ((unsigned)f2bf(a.w) << 16);
      pb.x = (unsigned)f2bf(b.x) | ((unsigned)f2bf(b.y) << 16);
      pb.y = (unsigned)f2bf(b.z) | ((unsigned)f2bf(b.w) << 16);
      *(uint2*)(&As[row * 32 + c4]) = pa;
      *(uint2*)(&Bs[row * 32 + c4]) = pb;
    }
    __syncthreads();

    bf16x16 af0 = load_frag_a(&As[(wm + 0) * 32], 32, lane);
    bf16x16 af1 = load_frag_a(&As[(wm + 16) * 32], 32, lane);
#pragma unroll
    for (int nt = 0; nt < 4; ++nt) {
      bf16x16 bf = load_frag_b(Bs, 32, lane, wn + nt * 16, 0);
      acc[0][nt] = wmma_bf16(af0, bf, acc[0][nt]);
      acc[1][nt] = wmma_bf16(af1, bf, acc[1][nt]);
    }
  }

#pragma unroll
  for (int nt = 0; nt < 4; ++nt) {
    int col = n0 + wn + nt * 16 + (lane & 15);
    float bb = bias[col];
#pragma unroll
    for (int mt = 0; mt < 2; ++mt) {
      int rowb = m0 + wm + mt * 16 + ((lane >> 4) * 8);
#pragma unroll
      for (int i = 0; i < 8; ++i) {
        float v = acc[mt][nt][i] + bb;
        v = v > 0.0f ? v : 0.0f;
        Out[(size_t)(rowb + i) * 1024 + col] = f2bf(v);
      }
    }
  }
}

// ---------------------------------------------------------------------------
// Flash attention: one block = (batch, head, 64-query tile); 4 waves x 16 rows.
// Q/K tiles staged with async global->LDS; online softmax in registers;
// post-softmax query mask; residual add.
// ---------------------------------------------------------------------------
__global__ __launch_bounds__(128) void flash_attn(
    const unsigned short* __restrict__ qp, const unsigned short* __restrict__ kp,
    const unsigned short* __restrict__ vp, const float* __restrict__ query,
    const float* __restrict__ masks, float* __restrict__ out) {
  __shared__ unsigned short Qs[64 * 64];
  __shared__ unsigned short Ks[64 * 64];
  __shared__ unsigned short Vt[64 * 72];     // V transposed [d][key], pad->16B rows
  __shared__ unsigned short Ps[4][16 * 64];  // per-wave P staging

  const int tid  = threadIdx.x;
  const int lane = tid & 31;
  const int wave = tid >> 5;
  const int qb = blockIdx.x * 64;   // query tile base
  const int hc = blockIdx.y * 64;   // head column base
  const int b  = blockIdx.z;
  const size_t rowBase = (size_t)b * 1024;

  // Q tile -> LDS (async, byte-identical copy)
#pragma unroll
  for (int r = 0; r < 4; ++r) {
    int j = tid + 128 * r;
    int row = j >> 3;
    int c8 = (j & 7) * 8;
    async_copy_b128(qp + (rowBase + qb + row) * 1024 + hc + c8, &Qs[row * 64 + c8]);
  }
  wait_async0();
  __syncthreads();

  bf16x16 qf0 = load_frag_a(&Qs[(wave * 16) * 64], 64, lane);       // d 0..31
  bf16x16 qf1 = load_frag_a(&Qs[(wave * 16) * 64 + 32], 64, lane);  // d 32..63

  const f32x8 zf = {0.f, 0.f, 0.f, 0.f, 0.f, 0.f, 0.f, 0.f};
  float mrow[8], lrow[8];
#pragma unroll
  for (int i = 0; i < 8; ++i) { mrow[i] = -1e30f; lrow[i] = 0.0f; }
  f32x8 O[4];
#pragma unroll
  for (int nt = 0; nt < 4; ++nt) O[nt] = zf;

  const float scale = 0.125f;  // 1/sqrt(64)

  for (int kt = 0; kt < 16; ++kt) {
    const int kb = kt * 64;
    __syncthreads();
    // K tile row-major via async copy; V tile transposed into Vt[d][key] manually
#pragma unroll
    for (int r = 0; r < 4; ++r) {
      int j = tid + 128 * r;
      int row = j >> 3;        // key index
      int c8 = (j & 7) * 8;    // d base
      async_copy_b128(kp + (rowBase + kb + row) * 1024 + hc + c8, &Ks[row * 64 + c8]);
      uint4 vv = *(const uint4*)(vp + (rowBase + kb + row) * 1024 + hc + c8);
      unsigned e0 = vv.x & 0xffff, e1 = vv.x >> 16, e2 = vv.y & 0xffff, e3 = vv.y >> 16;
      unsigned e4 = vv.z & 0xffff, e5 = vv.z >> 16, e6 = vv.w & 0xffff, e7 = vv.w >> 16;
      Vt[(c8 + 0) * 72 + row] = (unsigned short)e0;
      Vt[(c8 + 1) * 72 + row] = (unsigned short)e1;
      Vt[(c8 + 2) * 72 + row] = (unsigned short)e2;
      Vt[(c8 + 3) * 72 + row] = (unsigned short)e3;
      Vt[(c8 + 4) * 72 + row] = (unsigned short)e4;
      Vt[(c8 + 5) * 72 + row] = (unsigned short)e5;
      Vt[(c8 + 6) * 72 + row] = (unsigned short)e6;
      Vt[(c8 + 7) * 72 + row] = (unsigned short)e7;
    }
    wait_async0();
    __syncthreads();

    // S = Q x K^T  (16x64 per wave)
    f32x8 s[4];
#pragma unroll
    for (int nt = 0; nt < 4; ++nt) {
      s[nt] = zf;
      bf16x16 kf0 = load_frag_b(Ks, 64, lane, nt * 16, 0);
      bf16x16 kf1 = load_frag_b(Ks, 64, lane, nt * 16, 32);
      s[nt] = wmma_bf16(qf0, kf0, s[nt]);
      s[nt] = wmma_bf16(qf1, kf1, s[nt]);
    }

    // online softmax: row stats per lane (rows live in fixed 16-lane groups;
    // xor masks 1/2/4/8 stay inside each 16-lane half of the wave32)
    float fscale[8];
#pragma unroll
    for (int i = 0; i < 8; ++i) {
      float v = fmaxf(fmaxf(s[0][i], s[1][i]), fmaxf(s[2][i], s[3][i])) * scale;
#pragma unroll
      for (int d = 1; d < 16; d <<= 1) v = fmaxf(v, __shfl_xor(v, d));
      float mn = fmaxf(mrow[i], v);
      fscale[i] = __expf(mrow[i] - mn);
      mrow[i] = mn;
    }
#pragma unroll
    for (int nt = 0; nt < 4; ++nt)
#pragma unroll
      for (int i = 0; i < 8; ++i)
        s[nt][i] = __expf(s[nt][i] * scale - mrow[i]);
#pragma unroll
    for (int i = 0; i < 8; ++i) {
      float r = (s[0][i] + s[1][i]) + (s[2][i] + s[3][i]);
#pragma unroll
      for (int d = 1; d < 16; d <<= 1) r += __shfl_xor(r, d);
      lrow[i] = lrow[i] * fscale[i] + r;
    }
#pragma unroll
    for (int nt = 0; nt < 4; ++nt)
#pragma unroll
      for (int i = 0; i < 8; ++i) O[nt][i] *= fscale[i];

    // P (C-frag layout) -> LDS bf16, re-read as A-frags
    unsigned short* pw = &Ps[wave][0];
#pragma unroll
    for (int nt = 0; nt < 4; ++nt)
#pragma unroll
      for (int i = 0; i < 8; ++i)
        pw[((lane >> 4) * 8 + i) * 64 + nt * 16 + (lane & 15)] = f2bf(s[nt][i]);
    __syncthreads();

    bf16x16 pf0 = load_frag_a(&Ps[wave][0], 64, lane);
    bf16x16 pf1 = load_frag_a(&Ps[wave][0] + 32, 64, lane);
#pragma unroll
    for (int nt = 0; nt < 4; ++nt) {
      bf16x16 vf0 = load_frag_b(Vt, 72, lane, nt * 16, 0);
      bf16x16 vf1 = load_frag_b(Vt, 72, lane, nt * 16, 32);
      O[nt] = wmma_bf16(pf0, vf0, O[nt]);
      O[nt] = wmma_bf16(pf1, vf1, O[nt]);
    }
  }

  // epilogue: 1/rowsum * mask[b,q]  + residual (fp32)
  const int qrow0 = qb + wave * 16 + ((lane >> 4) * 8);
  float rinv[8];
#pragma unroll
  for (int i = 0; i < 8; ++i)
    rinv[i] = masks[rowBase + qrow0 + i] / lrow[i];
#pragma unroll
  for (int nt = 0; nt < 4; ++nt) {
    int dcol = hc + nt * 16 + (lane & 15);
#pragma unroll
    for (int i = 0; i < 8; ++i) {
      size_t idx = (rowBase + qrow0 + i) * 1024 + dcol;
      out[idx] = O[nt][i] * rinv[i] + query[idx];
    }
  }
}

extern "C" void kernel_launch(void* const* d_in, const int* in_sizes, int n_in,
                              void* d_out, int out_size, void* d_ws, size_t ws_size,
                              hipStream_t stream) {
  const float* query = (const float*)d_in[0];
  const float* key   = (const float*)d_in[1];
  const float* value = (const float*)d_in[2];
  const float* masks = (const float*)d_in[3];
  const float* Wq = (const float*)d_in[4];
  const float* bq = (const float*)d_in[5];
  const float* Wk = (const float*)d_in[6];
  const float* bk = (const float*)d_in[7];
  const float* Wv = (const float*)d_in[8];
  const float* bv = (const float*)d_in[9];
  float* out = (float*)d_out;

  unsigned short* qp = (unsigned short*)d_ws;            // 4096x1024 bf16
  unsigned short* kp = qp + (size_t)4096 * 1024;
  unsigned short* vp = kp + (size_t)4096 * 1024;

  proj_relu_gemm<<<dim3(32, 8, 3), 256, 0, stream>>>(query, key, value,
                                                     Wq, bq, Wk, bk, Wv, bv,
                                                     qp, kp, vp);
  flash_attn<<<dim3(16, 16, 4), 128, 0, stream>>>(qp, kp, vp, query, masks, out);
}
